// TensorVMSplit_20968030339386
// MI455X (gfx1250) — matrix operand order repaired and checked by reading, not compiled, not run on hardware
//
#include <hip/hip_runtime.h>
#include <math.h>

#define G 192
#define NSAMP 662
#define NCHUNK 11           // ceil(662/64)
#define WTHRESH 1e-4f
#define DIST_SCALE 25.0f

typedef _Float16 half16 __attribute__((ext_vector_type(16)));
typedef float    float8 __attribute__((ext_vector_type(8)));

__device__ __forceinline__ int clampi(int v, int lo, int hi) {
    return v < lo ? lo : (v > hi ? hi : v);
}

// A-fragment (16x32 f16, M x K): lanes 0-15 -> M=lane, K = k0..k0+15
//                                lanes 16-31 -> M=lane-16, K = k0+16..k0+31
__device__ __forceinline__ half16 frag_a(const _Float16* base, int lda, int m0, int k0, int lane) {
    int m  = m0 + (lane & 15);
    int kb = k0 + ((lane >> 4) << 4);
    const _Float16* p = base + m * lda + kb;
    half16 a;
#pragma unroll
    for (int j = 0; j < 16; ++j) a[j] = p[j];
    return a;
}

// B-fragment (32x16 f16, K x N) from row-major W[N][K]: lane holds column n=lane&15,
// K = k0 + (lane>=16 ? 16 : 0) .. +15  (16 contiguous K values per lane)
__device__ __forceinline__ half16 frag_b(const _Float16* base, int ldb, int n0, int k0, int lane) {
    int n  = n0 + (lane & 15);
    int kb = k0 + ((lane >> 4) << 4);
    const _Float16* p = base + n * ldb + kb;
    half16 b;
#pragma unroll
    for (int j = 0; j < 16; ++j) b[j] = p[j];
    return b;
}

// sum_c bilerp(plane_c,u,v) * lerp(line_c,l) over 16 channels
__device__ __forceinline__ float plane_line_dot(const float* __restrict__ plane,
                                                const float* __restrict__ line,
                                                float u, float v, float l) {
    float x = (u + 1.f) * 0.5f * (G - 1);
    float y = (v + 1.f) * 0.5f * (G - 1);
    float x0f = floorf(x), y0f = floorf(y);
    float wx = x - x0f, wy = y - y0f;
    int x0 = clampi((int)x0f, 0, G - 1), x1 = clampi((int)x0f + 1, 0, G - 1);
    int y0 = clampi((int)y0f, 0, G - 1), y1 = clampi((int)y0f + 1, 0, G - 1);
    float w00 = (1.f - wx) * (1.f - wy), w01 = wx * (1.f - wy);
    float w10 = (1.f - wx) * wy,         w11 = wx * wy;
    float zl = (l + 1.f) * 0.5f * (G - 1);
    float z0f = floorf(zl);
    float wz = zl - z0f;
    int z0 = clampi((int)z0f, 0, G - 1), z1 = clampi((int)z0f + 1, 0, G - 1);
    float acc = 0.f;
#pragma unroll 4
    for (int c = 0; c < 16; ++c) {
        const float* P = plane + c * G * G;
        float pv = w00 * P[y0 * G + x0] + w01 * P[y0 * G + x1] +
                   w10 * P[y1 * G + x0] + w11 * P[y1 * G + x1];
        const float* L = line + c * G;
        float lv = (1.f - wz) * L[z0] + wz * L[z1];
        acc += pv * lv;
    }
    return acc;
}

// per-channel coefs (not summed)
__device__ __forceinline__ void color_coef(const float* __restrict__ plane,
                                           const float* __restrict__ line,
                                           float u, float v, float l, float* out16) {
    float x = (u + 1.f) * 0.5f * (G - 1);
    float y = (v + 1.f) * 0.5f * (G - 1);
    float x0f = floorf(x), y0f = floorf(y);
    float wx = x - x0f, wy = y - y0f;
    int x0 = clampi((int)x0f, 0, G - 1), x1 = clampi((int)x0f + 1, 0, G - 1);
    int y0 = clampi((int)y0f, 0, G - 1), y1 = clampi((int)y0f + 1, 0, G - 1);
    float w00 = (1.f - wx) * (1.f - wy), w01 = wx * (1.f - wy);
    float w10 = (1.f - wx) * wy,         w11 = wx * wy;
    float zl = (l + 1.f) * 0.5f * (G - 1);
    float z0f = floorf(zl);
    float wz = zl - z0f;
    int z0 = clampi((int)z0f, 0, G - 1), z1 = clampi((int)z0f + 1, 0, G - 1);
#pragma unroll 4
    for (int c = 0; c < 16; ++c) {
        const float* P = plane + c * G * G;
        float pv = w00 * P[y0 * G + x0] + w01 * P[y0 * G + x1] +
                   w10 * P[y1 * G + x0] + w11 * P[y1 * G + x1];
        const float* L = line + c * G;
        float lv = (1.f - wz) * L[z0] + wz * L[z1];
        out16[c] = pv * lv;
    }
}

__global__ __launch_bounds__(128, 1)
void TensorVMSplit_20968030339386_kernel(
    const float* __restrict__ rays,    const float* __restrict__ dplanes,
    const float* __restrict__ dlines,  const float* __restrict__ cplanes,
    const float* __restrict__ clines,  const float* __restrict__ basisW,
    const float* __restrict__ w1,      const float* __restrict__ b1,
    const float* __restrict__ w2,      const float* __restrict__ b2,
    const float* __restrict__ w3,      const float* __restrict__ b3,
    float* __restrict__ out, int nrays)
{
    __shared__ _Float16 w1h[128 * 160];   // 40 KB (K padded 150->160, zeros)
    __shared__ _Float16 w2h[128 * 128];   // 32 KB
    __shared__ _Float16 w3p[16 * 128];    //  4 KB (w3 padded N 3->16, zeros)
    __shared__ float    b1s[128], b2s[128], b3s[3];
    __shared__ float    basisS[27 * 48];
    __shared__ _Float16 As[64 * 160];     // 20 KB  (A tile; reused as H2 64x128)
    __shared__ _Float16 H1[64 * 128];     // 16 KB
    __shared__ float    alphaS[NSAMP];
    __shared__ float    weightS[NSAMP];
    __shared__ float    rgbc[192];
    __shared__ float    redbuf[192];
    __shared__ float    wsumS, wtsumS;

    const int tid  = threadIdx.x;
    const int lane = tid & 31;
    const int wv   = tid >> 5;
    const int ray  = blockIdx.x;
    const float STEPF = (3.0f / 191.0f) * 0.5f;

    // ---- stage weights into LDS (f16) ----
    for (int i = tid; i < 128 * 150; i += 128)
        w1h[(i / 150) * 160 + (i % 150)] = (_Float16)w1[i];
    for (int i = tid; i < 128 * 10; i += 128)
        w1h[(i / 10) * 160 + 150 + (i % 10)] = (_Float16)0.f;
    for (int i = tid; i < 128 * 128; i += 128) w2h[i] = (_Float16)w2[i];
    for (int i = tid; i < 16 * 128; i += 128) {
        int n = i >> 7;
        w3p[i] = (n < 3) ? (_Float16)w3[i] : (_Float16)0.f;
    }
    for (int i = tid; i < 27 * 48; i += 128) basisS[i] = basisW[i];
    b1s[tid] = b1[tid];
    b2s[tid] = b2[tid];
    if (tid < 3) b3s[tid] = b3[tid];

    // ---- ray setup (each thread computes identically) ----
    float ox = rays[ray * 6 + 0], oy = rays[ray * 6 + 1], oz = rays[ray * 6 + 2];
    float dx = rays[ray * 6 + 3], dy = rays[ray * 6 + 4], dz = rays[ray * 6 + 5];
    float vx = (dx == 0.f) ? 1e-6f : dx;
    float vy = (dy == 0.f) ? 1e-6f : dy;
    float vz = (dz == 0.f) ? 1e-6f : dz;
    float m0 = fminf((1.5f - ox) / vx, (-1.5f - ox) / vx);
    float m1 = fminf((1.5f - oy) / vy, (-1.5f - oy) / vy);
    float m2 = fminf((1.5f - oz) / vz, (-1.5f - oz) / vz);
    float tminv = fminf(fmaxf(fmaxf(fmaxf(m0, m1), m2), 2.f), 6.f);
    __syncthreads();

    // ---- density -> alpha for all samples ----
    for (int s = tid; s < NSAMP; s += 128) {
        float t = tminv + STEPF * s;
        float px = ox + dx * t, py = oy + dy * t, pz = oz + dz * t;
        bool valid = (px >= -1.5f) && (px <= 1.5f) && (py >= -1.5f) &&
                     (py <= 1.5f) && (pz >= -1.5f) && (pz <= 1.5f);
        float alpha = 0.f;
        if (valid) {
            float xnx = (px + 1.5f) * (2.f / 3.f) - 1.f;
            float xny = (py + 1.5f) * (2.f / 3.f) - 1.f;
            float xnz = (pz + 1.5f) * (2.f / 3.f) - 1.f;
            float sig = 0.f;
            sig += plane_line_dot(dplanes + 0 * 16 * G * G, dlines + 0 * 16 * G, xnx, xny, xnz);
            sig += plane_line_dot(dplanes + 1 * 16 * G * G, dlines + 1 * 16 * G, xnx, xnz, xny);
            sig += plane_line_dot(dplanes + 2 * 16 * G * G, dlines + 2 * 16 * G, xny, xnz, xnx);
            float xarg  = sig - 10.f;
            float sigma = fmaxf(xarg, 0.f) + log1pf(__expf(-fabsf(xarg)));
            float delta = (s == NSAMP - 1) ? 0.f : STEPF;
            alpha = 1.f - __expf(-sigma * delta * DIST_SCALE);
        }
        alphaS[s] = alpha;
    }
    __syncthreads();

    // ---- serial transmittance scan (cheap vs MLP) ----
    if (tid == 0) {
        float T = 1.f, ws = 0.f, wts = 0.f;
        for (int s = 0; s < NSAMP; ++s) {
            float a = alphaS[s];
            float w = a * T;
            weightS[s] = w;
            ws  += w;
            wts += w * (tminv + STEPF * s);
            T *= (1.f - a + 1e-10f);
        }
        wsumS = ws;
        wtsumS = wts;
    }
    __syncthreads();

    float accR = 0.f, accG = 0.f, accB = 0.f;

    // ---- chunks of 64 samples through the WMMA MLP ----
    for (int chunk = 0; chunk < NCHUNK; ++chunk) {
        int s0 = chunk * 64;

        // build A tile [64][160] f16
        if (tid < 64) {
            int gs = s0 + tid;
            _Float16* arow = &As[tid * 160];
            bool live = (gs < NSAMP) && (weightS[gs] > WTHRESH);
            if (live) {
                float t = tminv + STEPF * gs;
                float px = ox + dx * t, py = oy + dy * t, pz = oz + dz * t;
                float xnx = (px + 1.5f) * (2.f / 3.f) - 1.f;
                float xny = (py + 1.5f) * (2.f / 3.f) - 1.f;
                float xnz = (pz + 1.5f) * (2.f / 3.f) - 1.f;
                float coef[48];
                color_coef(cplanes + 0 * 16 * G * G, clines + 0 * 16 * G, xnx, xny, xnz, &coef[0]);
                color_coef(cplanes + 1 * 16 * G * G, clines + 1 * 16 * G, xnx, xnz, xny, &coef[16]);
                color_coef(cplanes + 2 * 16 * G * G, clines + 2 * 16 * G, xny, xnz, xnx, &coef[32]);
                float feat[27];
                for (int j = 0; j < 27; ++j) {
                    float sum = 0.f;
#pragma unroll
                    for (int k = 0; k < 48; ++k) sum += basisS[j * 48 + k] * coef[k];
                    feat[j] = sum;
                    arow[j] = (_Float16)sum;
                }
                arow[27] = (_Float16)dx; arow[28] = (_Float16)dy; arow[29] = (_Float16)dz;
#pragma unroll
                for (int i = 0; i < 27; ++i) {
                    float f = feat[i];
                    arow[30 + 2 * i] = (_Float16)__sinf(f);
                    arow[31 + 2 * i] = (_Float16)__sinf(2.f * f);
                    arow[84 + 2 * i] = (_Float16)__cosf(f);
                    arow[85 + 2 * i] = (_Float16)__cosf(2.f * f);
                }
                float rdv[3] = {dx, dy, dz};
#pragma unroll
                for (int i = 0; i < 3; ++i) {
                    arow[138 + 2 * i] = (_Float16)__sinf(rdv[i]);
                    arow[139 + 2 * i] = (_Float16)__sinf(2.f * rdv[i]);
                    arow[144 + 2 * i] = (_Float16)__cosf(rdv[i]);
                    arow[145 + 2 * i] = (_Float16)__cosf(2.f * rdv[i]);
                }
#pragma unroll
                for (int j = 150; j < 160; ++j) arow[j] = (_Float16)0.f;
            } else {
#pragma unroll 8
                for (int j = 0; j < 160; ++j) arow[j] = (_Float16)0.f;
            }
        }
        __syncthreads();

        // layer 1: [64x160] @ [160x128] -> H1, relu   (32 tiles, 4 waves x 8)
        for (int i = 0; i < 8; ++i) {
            int tile = i * 4 + wv;
            int mt = (tile & 3) * 16;
            int nt = (tile >> 2) * 16;
            float8 acc = {};
            for (int k = 0; k < 160; k += 32) {
                half16 a = frag_a(As, 160, mt, k, lane);
                half16 b = frag_b(w1h, 160, nt, k, lane);
                acc = __builtin_amdgcn_wmma_f32_16x16x32_f16(false, a, false, b,
                                                             (short)0, acc, false, false);
            }
            int n  = nt + (lane & 15);
            float bias = b1s[n];
            int mb = mt + ((lane >> 4) << 3);
#pragma unroll
            for (int r = 0; r < 8; ++r) {
                float v = acc[r] + bias;
                H1[(mb + r) * 128 + n] = (_Float16)fmaxf(v, 0.f);
            }
        }
        __syncthreads();

        // layer 2: [64x128] @ [128x128] -> H2 (reuse As), relu
        _Float16* H2 = As;
        for (int i = 0; i < 8; ++i) {
            int tile = i * 4 + wv;
            int mt = (tile & 3) * 16;
            int nt = (tile >> 2) * 16;
            float8 acc = {};
            for (int k = 0; k < 128; k += 32) {
                half16 a = frag_a(H1, 128, mt, k, lane);
                half16 b = frag_b(w2h, 128, nt, k, lane);
                acc = __builtin_amdgcn_wmma_f32_16x16x32_f16(false, a, false, b,
                                                             (short)0, acc, false, false);
            }
            int n  = nt + (lane & 15);
            float bias = b2s[n];
            int mb = mt + ((lane >> 4) << 3);
#pragma unroll
            for (int r = 0; r < 8; ++r) {
                float v = acc[r] + bias;
                H2[(mb + r) * 128 + n] = (_Float16)fmaxf(v, 0.f);
            }
        }
        __syncthreads();

        // layer 3 via WMMA: [64x128] @ [128x16] (w3 padded N 3->16), one M-tile per wave
        {
            int mt = wv * 16;
            float8 acc = {};
            for (int k = 0; k < 128; k += 32) {
                half16 a = frag_a(H2, 128, mt, k, lane);
                half16 b = frag_b(w3p, 128, 0, k, lane);
                acc = __builtin_amdgcn_wmma_f32_16x16x32_f16(false, a, false, b,
                                                             (short)0, acc, false, false);
            }
            int n  = lane & 15;
            int mb = mt + ((lane >> 4) << 3);
            if (n < 3) {
                float bias = b3s[n];
#pragma unroll
                for (int r = 0; r < 8; ++r) {
                    float v = acc[r] + bias;
                    rgbc[(mb + r) * 3 + n] = 1.f / (1.f + __expf(-v));
                }
            }
        }
        __syncthreads();

        // weighted accumulation (masked, as in reference)
        if (tid < 64) {
            int gs = s0 + tid;
            if (gs < NSAMP) {
                float w = weightS[gs];
                if (w > WTHRESH) {
                    accR += w * rgbc[tid * 3 + 0];
                    accG += w * rgbc[tid * 3 + 1];
                    accB += w * rgbc[tid * 3 + 2];
                }
            }
        }
        __syncthreads();
    }

    // ---- reduce 64 partials, write outputs ----
    if (tid < 64) {
        redbuf[tid]       = accR;
        redbuf[64 + tid]  = accG;
        redbuf[128 + tid] = accB;
    }
    __syncthreads();
    for (int off = 32; off > 0; off >>= 1) {
        if (tid < off) {
            redbuf[tid]       += redbuf[tid + off];
            redbuf[64 + tid]  += redbuf[64 + tid + off];
            redbuf[128 + tid] += redbuf[128 + tid + off];
        }
        __syncthreads();
    }
    if (tid == 0) {
        float bg = 1.f - wsumS;
        out[ray * 3 + 0] = fminf(fmaxf(redbuf[0]   + bg, 0.f), 1.f);
        out[ray * 3 + 1] = fminf(fmaxf(redbuf[64]  + bg, 0.f), 1.f);
        out[ray * 3 + 2] = fminf(fmaxf(redbuf[128] + bg, 0.f), 1.f);
        out[nrays * 3 + ray] = wtsumS + bg * dz;   // depth: + (1-acc)*rays_chunk[:,-1]
    }
}

extern "C" void kernel_launch(void* const* d_in, const int* in_sizes, int n_in,
                              void* d_out, int out_size, void* d_ws, size_t ws_size,
                              hipStream_t stream) {
    const float* rays    = (const float*)d_in[0];
    const float* dplanes = (const float*)d_in[1];
    const float* dlines  = (const float*)d_in[2];
    const float* cplanes = (const float*)d_in[3];
    const float* clines  = (const float*)d_in[4];
    const float* basisW  = (const float*)d_in[5];
    const float* w1      = (const float*)d_in[6];
    const float* b1      = (const float*)d_in[7];
    const float* w2      = (const float*)d_in[8];
    const float* b2      = (const float*)d_in[9];
    const float* w3      = (const float*)d_in[10];
    const float* b3      = (const float*)d_in[11];
    int nrays = in_sizes[0] / 6;

    hipLaunchKernelGGL(TensorVMSplit_20968030339386_kernel,
                       dim3(nrays), dim3(128), 0, stream,
                       rays, dplanes, dlines, cplanes, clines, basisW,
                       w1, b1, w2, b2, w3, b3, (float*)d_out, nrays);
}
